// ABGuidanceNetwork_13091060318902
// MI455X (gfx1250) — matrix operand order repaired and checked by reading, not compiled, not run on hardware
//
#include <hip/hip_runtime.h>
#include <hip/hip_bf16.h>
#include <math.h>

// ---------------- CDNA5 WMMA types ----------------
typedef __attribute__((ext_vector_type(16))) _Float16 v16h;
typedef __attribute__((ext_vector_type(8)))  float    v8f;

// A-matrix (16x32 f16, MxK) per-lane K index for f16 slot s, lane-half h.
__device__ __forceinline__ int a_kmap(int s, int h) {
    return (s < 8) ? (h * 8 + s) : (16 + h * 8 + (s - 8));
}

#define WMMA_F16(A, B, C) \
    __builtin_amdgcn_wmma_f32_16x16x32_f16(false, (A), false, (B), (short)0, (C), false, false)

struct Ptr4 { const float* p[4]; };

// =====================================================================
// Kernel 1: projection  pr[lvl][n][o][pix] = W[lvl] @ x[lvl][n] + b[lvl]
//   GEMM 32x384 @ 384x1024 per (lvl,n); one wave per 32x16 output tile.
//   1024 waves total. Level weight slice (32x384) staged once per block
//   into LDS as f16 (24 KB of the WGP's 320 KB) -- shared by all 8 waves.
// =====================================================================
__global__ void __launch_bounds__(256)
proj_wmma_kernel(Ptr4 xs, const float* __restrict__ pw,
                 const float* __restrict__ pb, float* __restrict__ pr) {
    __shared__ _Float16 wlds[32 * 384];            // [o][c], 24 KB

    int tid  = threadIdx.x;
    int wave = (blockIdx.x * blockDim.x + tid) >> 5;
    int lane = tid & 31;
    int tile = wave & 63;          // 64 pixel tiles of 16
    int n    = (wave >> 6) & 3;
    int lvl  = (wave >> 8) & 3;    // block-uniform (8 waves differ in low 3 bits)

    // ---- stage f32 weights -> f16 LDS (coalesced, once per block) ----
    {
        const float* wsrc = pw + (size_t)lvl * 32 * 384;
        for (int i = tid; i < 32 * 384; i += 256)
            wlds[i] = (_Float16)wsrc[i];
    }
    __syncthreads();

    const float* x = xs.p[lvl];    // [4][384][1024]
    int p0 = tile * 16;
    int ln = lane & 15, lh = lane >> 4;

    v8f acc0 = {}, acc1 = {};
    for (int k0 = 0; k0 < 384; k0 += 32) {
        if (k0 + 32 < 384)
            __builtin_prefetch(&x[(size_t)(n * 384 + k0 + 32 + lh * 16) * 1024 + p0 + ln], 0, 3);
        v16h b;
#pragma unroll
        for (int s = 0; s < 16; ++s) {
            int c = k0 + lh * 16 + s;                       // B: K = h*16+s, N = ln
            b[s] = (_Float16)x[(size_t)(n * 384 + c) * 1024 + p0 + ln];
        }
        v16h a0, a1;
#pragma unroll
        for (int s = 0; s < 16; ++s) {
            int c = k0 + lh * 16 + s;
            a0[s] = wlds[ln * 384 + c];
            a1[s] = wlds[(16 + ln) * 384 + c];
        }
        acc0 = WMMA_F16(a0, b, acc0);
        acc1 = WMMA_F16(a1, b, acc1);
    }
#pragma unroll
    for (int r = 0; r < 8; ++r) {
        int o0 = r + lh * 8;
        int o1 = 16 + o0;
        size_t base = (size_t)((lvl * 4 + n) * 32);
        pr[(base + o0) * 1024 + p0 + ln] = acc0[r] + pb[lvl * 32 + o0];
        pr[(base + o1) * 1024 + p0 + ln] = acc1[r] + pb[lvl * 32 + o1];
    }
}

// =====================================================================
// Kernel 2: generic conv3x3 (32->32 ch, pad 1) as implicit GEMM.
//   K = 9 taps x 32 channels; one wave per 32x16 output tile.
//   Full 32x32x9 weight tensor staged per block into LDS as f16 (18 KB),
//   A-fragments gathered from LDS (ds_load) instead of redundant VMEM.
//   Optional epilogue: bias -> BN -> ReLU -> residual add.
// =====================================================================
__global__ void __launch_bounds__(256)
conv3_wmma_kernel(const float* __restrict__ x, const float* __restrict__ w,
                  const float* __restrict__ bias,
                  const float* __restrict__ bn_g, const float* __restrict__ bn_be,
                  const float* __restrict__ bn_m, const float* __restrict__ bn_v,
                  const float* __restrict__ add, float* __restrict__ out,
                  int H, int W, int relu_flag, int nWaves) {
    __shared__ _Float16 wlds[32 * 32 * 9];         // [(o*32+c)*9+tap], 18 KB

    int tid = threadIdx.x;
    for (int i = tid; i < 32 * 32 * 9; i += 256)
        wlds[i] = (_Float16)w[i];
    __syncthreads();

    int wave = (blockIdx.x * blockDim.x + tid) >> 5;
    if (wave >= nWaves) return;                 // wave-uniform: EXEC stays all-1s
    int lane = tid & 31;
    int xtiles = W >> 4;
    int t = wave % xtiles;
    int y = (wave / xtiles) % H;
    int n = wave / (xtiles * H);
    int ln = lane & 15, lh = lane >> 4;
    int px = t * 16 + ln;

    v8f acc0 = {}, acc1 = {};
    for (int ty = 0; ty < 3; ++ty) {
        int yy = y + ty - 1;
        bool yok = (yy >= 0) && (yy < H);
        for (int tx = 0; tx < 3; ++tx) {
            v16h b;
            int xx = px + tx - 1;
            bool xok = (xx >= 0) && (xx < W);
#pragma unroll
            for (int s = 0; s < 16; ++s) {
                int c = lh * 16 + s;
                float v = 0.f;
                if (yok && xok)
                    v = x[(size_t)((n * 32 + c) * H + yy) * W + xx];
                b[s] = (_Float16)v;
            }
            v16h a0, a1;
            int tap = ty * 3 + tx;
#pragma unroll
            for (int s = 0; s < 16; ++s) {
                int c = lh * 16 + s;
                a0[s] = wlds[(ln * 32 + c) * 9 + tap];
                a1[s] = wlds[((16 + ln) * 32 + c) * 9 + tap];
            }
            acc0 = WMMA_F16(a0, b, acc0);
            acc1 = WMMA_F16(a1, b, acc1);
        }
    }
#pragma unroll
    for (int r = 0; r < 8; ++r) {
        for (int half = 0; half < 2; ++half) {
            int o = half * 16 + r + lh * 8;
            float v = (half ? acc1[r] : acc0[r]) + bias[o];
            if (bn_g) {
                float inv = bn_g[o] * rsqrtf(bn_v[o] + 1e-5f);
                v = v * inv + (bn_be[o] - bn_m[o] * inv);
            }
            if (relu_flag) v = fmaxf(v, 0.f);
            size_t idx = (size_t)((n * 32 + o) * H + y) * W + t * 16 + ln;
            if (add) v += add[idx];
            out[idx] = v;
        }
    }
}

// =====================================================================
// Kernel 3: ConvTranspose2d k=4, s=2, p=1 (32->32ch), torch semantics.
//   out[o,oy,ox] = sum_{c,ky,kx} in[c,(oy+1-ky)/2,(ox+1-kx)/2] * w[c][o][ky][kx]
//   Parity -> only 2x2 taps per output. Direct VALU (tiny FLOPs, L2-resident).
// =====================================================================
__global__ void __launch_bounds__(256)
ct4_kernel(const float* __restrict__ x, const float* __restrict__ w,
           const float* __restrict__ bias,
           const float* __restrict__ g, const float* __restrict__ be,
           const float* __restrict__ m, const float* __restrict__ v,
           float* __restrict__ out, int Hin, int relu_flag) {
    int Hout = Hin * 2;
    int idx = blockIdx.x * blockDim.x + threadIdx.x;
    int total = 4 * 32 * Hout * Hout;
    if (idx >= total) return;
    int ox = idx % Hout;
    int oy = (idx / Hout) % Hout;
    int o  = (idx / (Hout * Hout)) % 32;
    int n  =  idx / (Hout * Hout * 32);

    float acc = bias[o];
    for (int ky = (oy + 1) & 1; ky < 4; ky += 2) {
        int iy = (oy + 1 - ky) >> 1;
        if (iy < 0 || iy >= Hin) continue;
        for (int kx = (ox + 1) & 1; kx < 4; kx += 2) {
            int ix = (ox + 1 - kx) >> 1;
            if (ix < 0 || ix >= Hin) continue;
            const float* xp = x + ((size_t)(n * 32) * Hin + iy) * Hin + ix; // +c*Hin*Hin
            const float* wp = w + (size_t)o * 16 + ky * 4 + kx;            // +c*512
#pragma unroll 8
            for (int c = 0; c < 32; ++c)
                acc += xp[(size_t)c * Hin * Hin] * wp[(size_t)c * 512];
        }
    }
    if (g) {
        float inv = g[o] * rsqrtf(v[o] + 1e-5f);
        acc = acc * inv + (be[o] - m[o] * inv);
    }
    if (relu_flag) acc = fmaxf(acc, 0.f);
    out[idx] = acc;
}

// =====================================================================
// Kernel 4: elementwise add (pr3 + pr2)
// =====================================================================
__global__ void add2_kernel(const float* __restrict__ a, const float* __restrict__ b,
                            float* __restrict__ o, int nEl) {
    int i = blockIdx.x * blockDim.x + threadIdx.x;
    if (i < nEl) o[i] = a[i] + b[i];
}

// =====================================================================
// Kernel 5: 9x9 local cost volume with reflect pad, /sqrt(32).
//   One thread per (n,h,w): fa channels cached in 32 VGPRs, 81 dots.
// =====================================================================
__global__ void __launch_bounds__(256)
corr_kernel(const float* __restrict__ fa, const float* __restrict__ fb,
            float* __restrict__ out) {
    int idx = blockIdx.x * blockDim.x + threadIdx.x;
    if (idx >= 4 * 256 * 256) return;
    int wc = idx & 255;
    int hc = (idx >> 8) & 255;
    int n  = idx >> 16;
    const size_t cs = 65536;
    const float* fap = fa + (size_t)n * 32 * cs + hc * 256 + wc;
    float areg[32];
#pragma unroll
    for (int c = 0; c < 32; ++c) areg[c] = fap[(size_t)c * cs];
    const float inv_scale = 0.17677669529663687f; // 1/sqrt(32)
    for (int dy = 0; dy < 9; ++dy) {
        int yy = hc + dy - 4;
        yy = yy < 0 ? -yy : (yy > 255 ? 510 - yy : yy);  // np 'reflect'
        for (int dx = 0; dx < 9; ++dx) {
            int xx = wc + dx - 4;
            xx = xx < 0 ? -xx : (xx > 255 ? 510 - xx : xx);
            const float* fbp = fb + (size_t)n * 32 * cs + yy * 256 + xx;
            float acc = 0.f;
#pragma unroll
            for (int c = 0; c < 32; ++c) acc += areg[c] * fbp[(size_t)c * cs];
            out[((size_t)(n * 81) + dy * 9 + dx) * 65536 + hc * 256 + wc] = acc * inv_scale;
        }
    }
}

// =====================================================================
// Host launcher
// =====================================================================
extern "C" void kernel_launch(void* const* d_in, const int* in_sizes, int n_in,
                              void* d_out, int out_size, void* d_ws, size_t ws_size,
                              hipStream_t stream) {
    (void)in_sizes; (void)n_in; (void)out_size; (void)ws_size;
    // dict order: fa1,fb1,fa2,fb2,fa3,fb3,fa4,fb4, then params in dict order
    Ptr4 A, B;
    for (int i = 0; i < 4; ++i) {
        A.p[i] = (const float*)d_in[2 * i];
        B.p[i] = (const float*)d_in[2 * i + 1];
    }
    const float* proj_w = (const float*)d_in[8];
    const float* proj_b = (const float*)d_in[9];
    const float* fuse_w = (const float*)d_in[10];
    const float* fuse_b = (const float*)d_in[11];
    const float* fuse_g = (const float*)d_in[12];
    const float* fuse_be= (const float*)d_in[13];
    const float* fuse_m = (const float*)d_in[14];
    const float* fuse_v = (const float*)d_in[15];
    const float* ct_w   = (const float*)d_in[16];
    const float* ct_b   = (const float*)d_in[17];
    const float* up_g   = (const float*)d_in[18];
    const float* up_be  = (const float*)d_in[19];
    const float* up_m   = (const float*)d_in[20];
    const float* up_v   = (const float*)d_in[21];
    const float* out_w  = (const float*)d_in[22];
    const float* out_b  = (const float*)d_in[23];

    float* ws = (float*)d_ws;
    const size_t LVL = 4 * 32 * 1024;       // 131072 (one pyramid level, all N)
    float* prA   = ws;                       // 4*LVL
    float* prB   = prA + 4 * LVL;            // 4*LVL
    float* xa    = prB + 4 * LVL;            // LVL
    float* ya    = xa + LVL;                 // LVL
    float* xb    = ya + LVL;                 // LVL
    float* yb    = xb + LVL;                 // LVL
    float* s64   = yb + LVL;                 // 4*32*64*64   = 524288
    float* s128  = s64 + 524288;             // 4*32*128*128 = 2097152
    float* s256  = s128 + 2097152;           // 4*32*256*256 = 8388608
    float* faF   = s256 + 8388608;           // 8388608
    float* fbF   = faF + 8388608;            // 8388608  (total ~118 MB)

    // --- projections (WMMA GEMM), 1024 waves each ---
    proj_wmma_kernel<<<128, 256, 0, stream>>>(A, proj_w, proj_b, prA);
    proj_wmma_kernel<<<128, 256, 0, stream>>>(B, proj_w, proj_b, prB);

    // --- x = pr3 + pr2 ---
    add2_kernel<<<(int)(LVL + 255) / 256, 256, 0, stream>>>(prA + 3 * LVL, prA + 2 * LVL, xa, (int)LVL);
    add2_kernel<<<(int)(LVL + 255) / 256, 256, 0, stream>>>(prB + 3 * LVL, prB + 2 * LVL, xb, (int)LVL);

    // --- fuse stages (conv3x3 + BN + ReLU [+ residual]), 32x32, 256 waves ---
    auto fuse = [&](const float* in, float* out, int j, const float* add) {
        int nW = 4 * 32 * 2;                        // n * H * (W/16)
        conv3_wmma_kernel<<<nW * 32 / 256, 256, 0, stream>>>(
            in, fuse_w + (size_t)j * 32 * 32 * 9, fuse_b + j * 32,
            fuse_g + j * 32, fuse_be + j * 32, fuse_m + j * 32, fuse_v + j * 32,
            add, out, 32, 32, 1, nW);
    };
    fuse(xa, ya, 0, prA + 1 * LVL);
    fuse(ya, xa, 1, prA + 0 * LVL);
    fuse(xa, ya, 2, nullptr);
    fuse(xb, yb, 0, prB + 1 * LVL);
    fuse(yb, xb, 1, prB + 0 * LVL);
    fuse(xb, yb, 2, nullptr);

    // --- decoder: 3x convT4x4 s2 (+BN/ReLU on first two) + out conv3x3 ---
    auto cascade_up = [&](const float* x32, float* fin) {
        int t0 = 4 * 32 * 64 * 64, t1 = 4 * 32 * 128 * 128, t2 = 4 * 32 * 256 * 256;
        ct4_kernel<<<(t0 + 255) / 256, 256, 0, stream>>>(
            x32, ct_w + 0 * 32 * 32 * 16, ct_b + 0, up_g + 0, up_be + 0, up_m + 0, up_v + 0,
            s64, 32, 1);
        ct4_kernel<<<(t1 + 255) / 256, 256, 0, stream>>>(
            s64, ct_w + 1 * 32 * 32 * 16, ct_b + 32, up_g + 32, up_be + 32, up_m + 32, up_v + 32,
            s128, 64, 1);
        ct4_kernel<<<(t2 + 255) / 256, 256, 0, stream>>>(
            s128, ct_w + 2 * 32 * 32 * 16, ct_b + 64,
            nullptr, nullptr, nullptr, nullptr, s256, 128, 0);
        int nW = 4 * 256 * 16;                      // n * H * (W/16) @ 256x256
        conv3_wmma_kernel<<<nW * 32 / 256, 256, 0, stream>>>(
            s256, out_w, out_b, nullptr, nullptr, nullptr, nullptr,
            nullptr, fin, 256, 256, 0, nW);
    };
    cascade_up(ya, faF);   // scratch buffers reused sequentially (same stream)
    cascade_up(yb, fbF);

    // --- 9x9 cost volume ---
    corr_kernel<<<(4 * 256 * 256) / 256, 256, 0, stream>>>(faF, fbF, (float*)d_out);
}
